// Hysteresis_50543175139674
// MI455X (gfx1250) — matrix operand order repaired and checked by reading, not compiled, not run on hardware
//
#include <hip/hip_runtime.h>
#include <math.h>

#define NMESH 256
#define MSTEPS 4096

typedef __attribute__((ext_vector_type(16))) _Float16     v16h;
typedef __attribute__((ext_vector_type(8)))  _Float16     v8h;
typedef __attribute__((ext_vector_type(8)))  float        v8f;
typedef __attribute__((ext_vector_type(2)))  short        s16x2;
typedef __attribute__((ext_vector_type(8)))  unsigned int u32x8;

// ---- workspace layout (bytes) ----
// [0,16)                : float sum (padded)
// [16, 16+131072)       : dens f16  (256*256*2)
// [131328, +2097152)    : U table  int16 (MSTEPS*NMESH)
// [2228480, +2097152)   : D table  int16
#define OFF_DENS 16
#define OFF_U    131328
#define OFF_D    2228480

// Kernel 1: softplus density, scatter into lower triangle, f32 total sum.
__global__ void __launch_bounds__(256) prep_density(const float* __restrict__ raw,
                                                    _Float16* __restrict__ dens,
                                                    float* __restrict__ sum) {
    int idx = blockIdx.x * 256 + threadIdx.x;   // 0..65535
    int i = idx >> 8, j = idx & 255;
    float d = 0.0f;
    if (j <= i) {
        float x = raw[(i * (i + 1)) / 2 + j];
        d = (x > 20.0f) ? x : log1pf(__expf(x));
    }
    dens[idx] = (_Float16)d;

    __shared__ float red[256];
    red[threadIdx.x] = d;
    __syncthreads();
    for (int s = 128; s > 0; s >>= 1) {
        if (threadIdx.x < s) red[threadIdx.x] += red[threadIdx.x + s];
        __syncthreads();
    }
    if (threadIdx.x == 0) atomicAdd(sum, red[0]);
}

// Kernel 2: build U/D last-event tables. Thread i owns grid level lin[i]
// (alpha_i for U, beta_i for D). Event 0 is the implicit up-sweep to 1.0
// (hs[0] = -H_MIN; reference's wrap makes step 0 an up-sweep since h<1).
__global__ void __launch_bounds__(256) build_ud(const float* __restrict__ h,
                                                const float* __restrict__ xx,   // row 0 = linspace
                                                short* __restrict__ U,
                                                short* __restrict__ D) {
    __shared__ float hl[MSTEPS];
    for (int t = threadIdx.x; t < MSTEPS; t += 256) hl[t] = h[t];
    __syncthreads();

    int i = threadIdx.x;
    float lv = xx[i];                      // lin[i], exact reference grid value
    short u = (1.0f > lv) ? (short)0 : (short)-1;   // event 0: up @ 1.0
    short d = -1;
    float hp = 1.0f;
    for (int tau = 1; tau <= MSTEPS; ++tau) {
        float hc = hl[tau - 1];
        if (hc > hp)      { if (hc > lv) u = (short)tau; }   // up: flips on alpha < hc
        else if (hc < hp) { if (hc < lv) d = (short)tau; }   // down: flips off beta > hc
        U[(tau - 1) * NMESH + i] = u;
        D[(tau - 1) * NMESH + i] = d;
        hp = hc;
    }
}

// sign(d - u) -> packed f16 {+1 if u>d else -1} for two i16 lanes at once.
__device__ __forceinline__ unsigned int sgn2(unsigned int dpair, s16x2 us2) {
    s16x2 d2 = __builtin_bit_cast(s16x2, dpair);
    s16x2 q  = d2 - us2;                               // v_pk_sub_i16
    unsigned int qw = __builtin_bit_cast(unsigned int, q);
    return (qw & 0x80008000u) ^ 0xBC00BC00u;           // sign->0x3C00 (+1), else 0xBC00 (-1)
}

__device__ __forceinline__ v16h mk_a(const uint4& G0, const uint4& G1, s16x2 us2) {
    u32x8 aw;
    aw[0] = sgn2(G0.x, us2);  aw[1] = sgn2(G0.y, us2);
    aw[2] = sgn2(G0.z, us2);  aw[3] = sgn2(G0.w, us2);
    aw[4] = sgn2(G1.x, us2);  aw[5] = sgn2(G1.y, us2);
    aw[6] = sgn2(G1.z, us2);  aw[7] = sgn2(G1.w, us2);
    return __builtin_bit_cast(v16h, aw);
}

// Kernel 3: b[t] = a * sum_{i,j} dens[i,j] * sign(U[t][i] - D[t][j])
// One block = one 16-timestep tile. 8 waves each own 32 alpha-rows.
// The per-lane D groups (invariant across rows) are hoisted into 16 uint4
// registers; the A state tile is generated with packed i16 sign math; two
// independent accumulator chains keep consecutive v_wmma_f32_16x16x32_f16
// ops overlapped with the VALU tile generation.
__global__ void __launch_bounds__(256) wmma_contract(const _Float16* __restrict__ dens,
                                                     const short* __restrict__ U,
                                                     const short* __restrict__ D,
                                                     const float* __restrict__ sumptr,
                                                     float* __restrict__ out) {
    constexpr int RS = 264;                 // padded LDS row stride (shorts): 528B = 132 dwords (no bank conflicts)
    __shared__ short    Ul[16 * RS];
    __shared__ short    Dl[16 * RS];
    __shared__ _Float16 Brow[8 * 256];      // per-wave dens row staging (4KB)
    __shared__ float    red[8 * 16];

    int tile = blockIdx.x;                  // 0..255 (16 timesteps each)
    int tid  = threadIdx.x;

    // Stage the 16 U rows / 16 D rows for this timestep tile into LDS.
    for (int r = 0; r < 16; ++r) {
        Ul[r * RS + tid] = U[(tile * 16 + r) * NMESH + tid];
        Dl[r * RS + tid] = D[(tile * 16 + r) * NMESH + tid];
    }
    __syncthreads();

    int wave  = tid >> 5;
    int lane  = tid & 31;
    int m     = lane & 15;                  // A/C row (timestep within tile)
    int kbase = (lane < 16) ? 0 : 8;        // A-matrix K base per lane half
    const short* Um  = &Ul[m * RS];
    _Float16*    brw = &Brow[wave * 256];

    // Hoist this lane's D groups for all 8 chunks: 16 x ds_load_b128, kept in VGPRs.
    const uint4* Dq = (const uint4*)(&Dl[m * RS]);     // 528B-aligned base -> 16B aligned
    uint4 G0[8], G1[8];
#pragma unroll
    for (int jc = 0; jc < 8; ++jc) {
        G0[jc] = Dq[(jc * 32 + kbase) >> 3];           // shorts [jb+kbase .. +7]
        G1[jc] = Dq[(jc * 32 + 16 + kbase) >> 3];      // shorts [jb+16+kbase .. +7]
    }

    v16h b0 = {}, b1 = {};                  // persistent: zero lanes stay zero forever
    v8f  acc0 = {}, acc1 = {};

    for (int ii = 0; ii < 32; ++ii) {
        int i = wave * 32 + ii;             // alpha row owned this iteration

        // Stage dens row i: 32 lanes x 16B = 512B, fully coalesced.
        ((v8h*)brw)[lane] = ((const v8h*)(dens + i * NMESH))[lane];

        short u = Um[i];                    // one DS load per row
        s16x2 us2 = {u, u};

#pragma unroll
        for (int jc = 0; jc < 8; jc += 2) {
            // ---- chunk jc -> acc0 ----
            {
                int jb = jc * 32;
                v16h a = mk_a(G0[jc], G1[jc], us2);
                if ((lane & 15) == 0) {
                    const _Float16* p = brw + jb + ((lane >> 4) << 3);
#pragma unroll
                    for (int e = 0; e < 16; ++e) b0[e] = p[(e & 7) + ((e >> 3) << 4)];
                }
                acc0 = __builtin_amdgcn_wmma_f32_16x16x32_f16(false, a, false, b0,
                                                              (short)0, acc0, false, false);
            }
            // ---- chunk jc+1 -> acc1 (independent chain) ----
            {
                int jb = (jc + 1) * 32;
                v16h a = mk_a(G0[jc + 1], G1[jc + 1], us2);
                if ((lane & 15) == 0) {
                    const _Float16* p = brw + jb + ((lane >> 4) << 3);
#pragma unroll
                    for (int e = 0; e < 16; ++e) b1[e] = p[(e & 7) + ((e >> 3) << 4)];
                }
                acc1 = __builtin_amdgcn_wmma_f32_16x16x32_f16(false, a, false, b1,
                                                              (short)0, acc1, false, false);
            }
        }
    }

    v8f acc;
#pragma unroll
    for (int r = 0; r < 8; ++r) acc[r] = acc0[r] + acc1[r];

    // Column 0 of C: lane 0 holds M=0..7, lane 16 holds M=8..15.
    if (lane == 0) {
#pragma unroll
        for (int r = 0; r < 8; ++r) red[wave * 16 + r] = acc[r];
    } else if (lane == 16) {
#pragma unroll
        for (int r = 0; r < 8; ++r) red[wave * 16 + 8 + r] = acc[r];
    }
    __syncthreads();

    if (tid < 16) {
        float s = 0.0f;
#pragma unroll
        for (int w = 0; w < 8; ++w) s += red[w * 16 + tid];
        out[tile * 16 + tid] = s * (1.0f / *sumptr);   // a = B_SAT / sum(dens)
    }
}

extern "C" void kernel_launch(void* const* d_in, const int* in_sizes, int n_in,
                              void* d_out, int out_size, void* d_ws, size_t ws_size,
                              hipStream_t stream) {
    const float* h   = (const float*)d_in[0];   // (M,)
    const float* raw = (const float*)d_in[1];   // (N*(N+1)/2,)
    const float* xx  = (const float*)d_in[2];   // (N,N), row 0 = linspace(-1,1,N)
    // d_in[3] = yy, same grid values; not needed.

    char*     ws    = (char*)d_ws;
    float*    d_sum = (float*)ws;
    _Float16* dens  = (_Float16*)(ws + OFF_DENS);
    short*    Utab  = (short*)(ws + OFF_U);
    short*    Dtab  = (short*)(ws + OFF_D);
    float*    outp  = (float*)d_out;

    hipMemsetAsync(d_sum, 0, sizeof(float), stream);
    prep_density<<<256, 256, 0, stream>>>(raw, dens, d_sum);
    build_ud<<<1, 256, 0, stream>>>(h, xx, Utab, Dtab);
    wmma_contract<<<256, 256, 0, stream>>>(dens, Utab, Dtab, d_sum, outp);
}
